// SpeechResModel_28621662061097
// MI455X (gfx1250) — compile-verified, matrix operand order checked by reading, not run on hardware
//
#include <hip/hip_runtime.h>
#include <stdint.h>
#include <stddef.h>

// ---------------------------------------------------------------------------
// Types for CDNA5 WMMA (wave32, v_wmma_f32_16x16x32_bf16)
// ---------------------------------------------------------------------------
typedef __bf16 bf16;
typedef __attribute__((ext_vector_type(16))) __bf16 v16bf;
typedef __attribute__((ext_vector_type(8)))  __bf16 v8bf;
typedef __attribute__((ext_vector_type(8)))  float  v8f;

// Model dims
#define TT    100            // time steps after conv
#define BB    512            // batch
#define HID   128
#define G4    512            // 4*HID
#define IN0   576            // 32 ch * 18 w
#define HIMG  812
#define WIMG  40
#define WOUT  18
#define NCLS  12

// ---------------------------------------------------------------------------
// Fragment load: 16 bf16 = two 16-byte chunks at p[0..7] and p[16..23]
// (A: row m, B: row n of B[n][k]; both contiguous in k) per ISA 7.12.2
// ---------------------------------------------------------------------------
__device__ __forceinline__ v16bf load_frag(const bf16* p) {
    v8bf lo = *(const v8bf*)(p);
    v8bf hi = *(const v8bf*)(p + 16);
    v16bf r;
#pragma unroll
    for (int e = 0; e < 8; ++e) { r[e] = lo[e]; r[e + 8] = hi[e]; }
    return r;
}

// ---------------------------------------------------------------------------
// Generic WMMA GEMM:  C[M,N] = A[M,K](bf16) * Bm[N,K]^T(bf16)  (+bias)(+add)
// One wave computes a 16 x (NT*16) strip; NT is a compile-time constant so
// there is NO divergent control flow around any WMMA (EXEC stays all-ones;
// the only branch is the scalar wave-bounds check via readfirstlane).
// Requires: M % 16 == 0, N % (NT*16) == 0, K % 32 == 0.
// Epilogue: +bias[n], +add[m][n], optional tanh; store f32 (Cf) or bf16 (Cb);
// only columns n < Nstore are written (divergent, but after all WMMAs).
// ---------------------------------------------------------------------------
template <int NT>
__global__ __launch_bounds__(256)
void gemm_wmma(const bf16* __restrict__ A, int lda,
               const bf16* __restrict__ Bm, int ldb,
               const float* __restrict__ bias,
               const float* __restrict__ add, int ldadd,
               float* __restrict__ Cf, bf16* __restrict__ Cb, int ldc,
               int M, int N, int K, int Nstore, int do_tanh) {
    // Scalarize the wave id so bounds checks are s_cbranch, not exec masks.
    const int wave = __builtin_amdgcn_readfirstlane(
        (int)((blockIdx.x * blockDim.x + threadIdx.x) >> 5));
    const int lane = threadIdx.x & 31;
    const int numM = M >> 4;
    const int numN = N / (NT * 16);
    if (wave >= numM * numN) return;           // scalar-uniform exit
    const int m0     = (wave % numM) << 4;
    const int n0base = (wave / numM) * (NT * 16);

    v8f acc[NT] = {};
    const int kb   = (lane >> 4) << 3;          // 0 or 8
    const int nl   = lane & 15;
    const bf16* arow = A + (size_t)(m0 + nl) * lda + kb;
    const bf16* brow0 = Bm + (size_t)(n0base + nl) * ldb + kb;

    for (int k0 = 0; k0 < K; k0 += 32) {
        v16bf afr = load_frag(arow + k0);
        v16bf bfr[NT];
#pragma unroll
        for (int j = 0; j < NT; ++j)
            bfr[j] = load_frag(brow0 + (size_t)(j * 16) * ldb + k0);
#pragma unroll
        for (int j = 0; j < NT; ++j)
            acc[j] = __builtin_amdgcn_wmma_f32_16x16x32_bf16(
                false, afr, false, bfr[j], (short)0, acc[j], false, false);
    }

    const int moff = (lane >> 4) << 3;          // 0 or 8
#pragma unroll
    for (int j = 0; j < NT; ++j) {
        int n = n0base + (j << 4) + nl;
        float bval = bias ? bias[n] : 0.0f;
#pragma unroll
        for (int r = 0; r < 8; ++r) {
            int m = m0 + moff + r;
            float v = acc[j][r] + bval;
            if (add)     v += add[(size_t)m * ldadd + n];
            if (do_tanh) v = tanhf(v);
            if (n < Nstore) {
                if (Cf) Cf[(size_t)m * ldc + n] = v;
                else    Cb[(size_t)m * ldc + n] = (bf16)v;
            }
        }
    }
}

// ---------------------------------------------------------------------------
// Conv2d(1->32,k=(20,5),s=(8,2)) + ReLU, writing seq[(t*B+b)][w*32+c] in bf16
// One thread per (b,t,w): loads 20x5 patch once, 3200 FMA vs LDS weights.
// ---------------------------------------------------------------------------
__global__ __launch_bounds__(256)
void conv_relu(const float* __restrict__ x, const float* __restrict__ cw,
               const float* __restrict__ cb, bf16* __restrict__ seq) {
    __shared__ float wsh[3200];
    __shared__ float bsh[32];
    for (int i = threadIdx.x; i < 3200; i += blockDim.x) wsh[i] = cw[i];
    if (threadIdx.x < 32) bsh[threadIdx.x] = cb[threadIdx.x];
    __syncthreads();

    int idx = blockIdx.x * blockDim.x + threadIdx.x;
    if (idx >= BB * TT * WOUT) return;
    int w = idx % WOUT;
    int b = (idx / WOUT) % BB;
    int t = idx / (WOUT * BB);

    const float* xp = x + (size_t)b * (HIMG * WIMG) + (t * 8) * WIMG + (w * 2);
    float patch[100];
#pragma unroll
    for (int kh = 0; kh < 20; ++kh)
#pragma unroll
        for (int kw = 0; kw < 5; ++kw)
            patch[kh * 5 + kw] = xp[kh * WIMG + kw];

    bf16* out = seq + ((size_t)t * BB + b) * IN0 + w * 32;
#pragma unroll 1
    for (int c = 0; c < 32; ++c) {
        float acc = bsh[c];
        const float* wc = &wsh[c * 100];
#pragma unroll
        for (int i = 0; i < 100; ++i) acc = fmaf(patch[i], wc[i], acc);
        out[c] = (bf16)fmaxf(acc, 0.0f);
    }
}

// ---------------------------------------------------------------------------
// LSTM gate elementwise: g[b][0:512] (i,f,g,o), c state, h out (bf16)
// ---------------------------------------------------------------------------
__global__ __launch_bounds__(256)
void lstm_elem(const float* __restrict__ g, float* __restrict__ c,
               bf16* __restrict__ hout, int first) {
    int idx = blockIdx.x * blockDim.x + threadIdx.x;
    if (idx >= BB * HID) return;
    int b = idx >> 7, h = idx & (HID - 1);
    const float* gr = g + (size_t)b * G4;
    float gi = gr[h], gf = gr[HID + h], gg = gr[2 * HID + h], go = gr[3 * HID + h];
    float si = 1.0f / (1.0f + __expf(-gi));
    float sf = 1.0f / (1.0f + __expf(-gf));
    float so = 1.0f / (1.0f + __expf(-go));
    float cold = first ? 0.0f : c[idx];
    float cn = sf * cold + si * tanhf(gg);
    c[idx] = cn;
    hout[idx] = (bf16)(so * tanhf(cn));
}

// ---------------------------------------------------------------------------
// Attention score: scores[row] = sum_h squish[row][h] * v[h]   (wave per row)
// ---------------------------------------------------------------------------
__global__ __launch_bounds__(256)
void att_score(const float* __restrict__ squish, const float* __restrict__ v,
               float* __restrict__ scores) {
    int gtid = blockIdx.x * blockDim.x + threadIdx.x;
    int wave = gtid >> 5, lane = gtid & 31;
    if (wave >= TT * BB) return;
    const float* row = squish + (size_t)wave * HID;
    float s = 0.0f;
#pragma unroll
    for (int e = 0; e < 4; ++e) {
        int h = lane + 32 * e;
        s += row[h] * v[h];
    }
#pragma unroll
    for (int off = 16; off > 0; off >>= 1) s += __shfl_xor(s, off, 32);
    if (lane == 0) scores[wave] = s;
}

// ---------------------------------------------------------------------------
// Softmax over T (per batch) + weighted pooling of h2.  Block = 1 batch,
// 128 threads (one per hidden dim).
// ---------------------------------------------------------------------------
__global__ __launch_bounds__(128)
void att_pool(const float* __restrict__ scores, const bf16* __restrict__ h2,
              bf16* __restrict__ pooled) {
    __shared__ float sc[TT];
    __shared__ float red[2];
    int b = blockIdx.x, tid = threadIdx.x;
    if (tid < TT) sc[tid] = scores[tid * BB + b];
    __syncthreads();
    if (tid == 0) {
        float mx = sc[0];
        for (int t = 1; t < TT; ++t) mx = fmaxf(mx, sc[t]);
        red[0] = mx;
    }
    __syncthreads();
    float mx = red[0];
    if (tid < TT) sc[tid] = __expf(sc[tid] - mx);
    __syncthreads();
    if (tid == 0) {
        float s = 0.0f;
        for (int t = 0; t < TT; ++t) s += sc[t];
        red[1] = s;
    }
    __syncthreads();
    float inv = 1.0f / red[1];
    float acc = 0.0f;
    for (int t = 0; t < TT; ++t)
        acc += sc[t] * (float)h2[((size_t)t * BB + b) * HID + tid];
    pooled[(size_t)b * HID + tid] = (bf16)(acc * inv);
}

// ---------------------------------------------------------------------------
// Weight prep kernels
// ---------------------------------------------------------------------------
__global__ void cvt_bf16(bf16* dst, const float* src, int n) {
    int i = blockIdx.x * blockDim.x + threadIdx.x;
    if (i < n) dst[i] = (bf16)src[i];
}
__global__ void cvt_bf16_T(bf16* dst, const float* src, int rows, int cols) {
    // dst[c*rows + r] = src[r*cols + c]
    int i = blockIdx.x * blockDim.x + threadIdx.x;
    if (i >= rows * cols) return;
    int r = i / cols, c = i % cols;
    dst[(size_t)c * rows + r] = (bf16)src[i];
}
__global__ void vec_add(float* dst, const float* a, const float* b, int n) {
    int i = blockIdx.x * blockDim.x + threadIdx.x;
    if (i < n) dst[i] = a[i] + b[i];
}
__global__ void pad_w2(bf16* dst, const float* w2, float* bdst, const float* b2) {
    int i = blockIdx.x * blockDim.x + threadIdx.x;  // 16*64 threads
    if (i >= 16 * 64) return;
    int n = i / 64, k = i % 64;
    dst[i] = (n < NCLS) ? (bf16)w2[n * 64 + k] : (bf16)0.0f;
    if (i < 16) bdst[i] = (i < NCLS) ? b2[i] : 0.0f;
}

// ---------------------------------------------------------------------------
static inline void launch_gemm(const bf16* A, int lda, const bf16* Bm, int ldb,
                               const float* bias, const float* add, int ldadd,
                               float* Cf, bf16* Cb, int ldc,
                               int M, int N, int K, int Nstore, int do_tanh,
                               int nt, hipStream_t s) {
    int strip = nt * 16;
    int numM = M >> 4, numN = N / strip;
    int waves = numM * numN;
    int blocks = (waves + 7) / 8;              // 8 waves / 256-thread block
    if (nt == 4)
        hipLaunchKernelGGL((gemm_wmma<4>), dim3(blocks), dim3(256), 0, s,
                           A, lda, Bm, ldb, bias, add, ldadd, Cf, Cb, ldc,
                           M, N, K, Nstore, do_tanh);
    else
        hipLaunchKernelGGL((gemm_wmma<1>), dim3(blocks), dim3(256), 0, s,
                           A, lda, Bm, ldb, bias, add, ldadd, Cf, Cb, ldc,
                           M, N, K, Nstore, do_tanh);
}

extern "C" void kernel_launch(void* const* d_in, const int* in_sizes, int n_in,
                              void* d_out, int out_size, void* d_ws, size_t ws_size,
                              hipStream_t stream) {
    (void)in_sizes; (void)n_in; (void)out_size; (void)ws_size;
    const float* x      = (const float*)d_in[0];
    const float* conv_w = (const float*)d_in[1];
    const float* conv_b = (const float*)d_in[2];
    const float* Wih0   = (const float*)d_in[3];
    const float* Whh0   = (const float*)d_in[4];
    const float* bih0   = (const float*)d_in[5];
    const float* bhh0   = (const float*)d_in[6];
    const float* Wih1   = (const float*)d_in[7];
    const float* Whh1   = (const float*)d_in[8];
    const float* bih1   = (const float*)d_in[9];
    const float* bhh1   = (const float*)d_in[10];
    const float* W_att  = (const float*)d_in[11];
    const float* b_att  = (const float*)d_in[12];
    const float* v_att  = (const float*)d_in[13];
    const float* W1     = (const float*)d_in[14];
    const float* b1     = (const float*)d_in[15];
    const float* W2     = (const float*)d_in[16];
    const float* b2     = (const float*)d_in[17];

    // ---- workspace carve-up (bump allocator, 256B aligned) ----
    char* p = (char*)d_ws;
    auto alloc = [&](size_t bytes) -> char* {
        char* r = p;
        p += (bytes + 255) & ~(size_t)255;
        return r;
    };
    bf16* wih0 = (bf16*)alloc((size_t)G4 * IN0 * 2);
    bf16* whh0 = (bf16*)alloc((size_t)G4 * HID * 2);
    bf16* wih1 = (bf16*)alloc((size_t)G4 * HID * 2);
    bf16* whh1 = (bf16*)alloc((size_t)G4 * HID * 2);
    bf16* watt = (bf16*)alloc((size_t)HID * HID * 2);   // transposed W_att
    bf16* w1b  = (bf16*)alloc((size_t)64 * HID * 2);
    bf16* w2p  = (bf16*)alloc((size_t)16 * 64 * 2);     // padded to N=16
    float* b0s = (float*)alloc(G4 * 4);
    float* b1s = (float*)alloc(G4 * 4);
    float* b2p = (float*)alloc(16 * 4);
    bf16* seq    = (bf16*)alloc((size_t)TT * BB * IN0 * 2);   // 59 MB
    float* xp    = (float*)alloc((size_t)TT * BB * G4 * 4);   // 105 MB (reused as squish)
    bf16* hseq1  = (bf16*)alloc((size_t)TT * BB * HID * 2);   // 13 MB
    bf16* hseq2  = (bf16*)alloc((size_t)TT * BB * HID * 2);   // 13 MB
    float* cst   = (float*)alloc((size_t)BB * HID * 4);
    float* g     = (float*)alloc((size_t)BB * G4 * 4);
    float* scores= (float*)alloc((size_t)TT * BB * 4);
    bf16* pooled = (bf16*)alloc((size_t)BB * HID * 2);
    bf16* hdn    = (bf16*)alloc((size_t)BB * 64 * 2);

    // ---- weight conversion ----
    hipLaunchKernelGGL(cvt_bf16, dim3((G4 * IN0 + 255) / 256), dim3(256), 0, stream, wih0, Wih0, G4 * IN0);
    hipLaunchKernelGGL(cvt_bf16, dim3((G4 * HID + 255) / 256), dim3(256), 0, stream, whh0, Whh0, G4 * HID);
    hipLaunchKernelGGL(cvt_bf16, dim3((G4 * HID + 255) / 256), dim3(256), 0, stream, wih1, Wih1, G4 * HID);
    hipLaunchKernelGGL(cvt_bf16, dim3((G4 * HID + 255) / 256), dim3(256), 0, stream, whh1, Whh1, G4 * HID);
    hipLaunchKernelGGL(cvt_bf16_T, dim3((HID * HID + 255) / 256), dim3(256), 0, stream, watt, W_att, HID, HID);
    hipLaunchKernelGGL(cvt_bf16, dim3((64 * HID + 255) / 256), dim3(256), 0, stream, w1b, W1, 64 * HID);
    hipLaunchKernelGGL(pad_w2, dim3(4), dim3(256), 0, stream, w2p, W2, b2p, b2);
    hipLaunchKernelGGL(vec_add, dim3(2), dim3(256), 0, stream, b0s, bih0, bhh0, G4);
    hipLaunchKernelGGL(vec_add, dim3(2), dim3(256), 0, stream, b1s, bih1, bhh1, G4);

    // ---- conv + relu -> seq (bf16, GEMM layout) ----
    hipLaunchKernelGGL(conv_relu, dim3((BB * TT * WOUT + 255) / 256), dim3(256), 0, stream,
                       x, conv_w, conv_b, seq);

    // ---- layer 0: xp = seq @ Wih0^T + (bih0+bhh0) ----
    launch_gemm(seq, IN0, wih0, IN0, b0s, nullptr, 0, xp, nullptr, G4,
                TT * BB, G4, IN0, G4, 0, 4, stream);
    for (int t = 0; t < TT; ++t) {
        const float* gt = xp + (size_t)t * BB * G4;
        if (t > 0) {
            launch_gemm(hseq1 + (size_t)(t - 1) * BB * HID, HID, whh0, HID,
                        nullptr, gt, G4, g, nullptr, G4,
                        BB, G4, HID, G4, 0, 4, stream);
            gt = g;
        }
        hipLaunchKernelGGL(lstm_elem, dim3(BB * HID / 256), dim3(256), 0, stream,
                           gt, cst, hseq1 + (size_t)t * BB * HID, (t == 0) ? 1 : 0);
    }

    // ---- layer 1: xp = hseq1 @ Wih1^T + (bih1+bhh1) ----
    launch_gemm(hseq1, HID, wih1, HID, b1s, nullptr, 0, xp, nullptr, G4,
                TT * BB, G4, HID, G4, 0, 4, stream);
    for (int t = 0; t < TT; ++t) {
        const float* gt = xp + (size_t)t * BB * G4;
        if (t > 0) {
            launch_gemm(hseq2 + (size_t)(t - 1) * BB * HID, HID, whh1, HID,
                        nullptr, gt, G4, g, nullptr, G4,
                        BB, G4, HID, G4, 0, 4, stream);
            gt = g;
        }
        hipLaunchKernelGGL(lstm_elem, dim3(BB * HID / 256), dim3(256), 0, stream,
                           gt, cst, hseq2 + (size_t)t * BB * HID, (t == 0) ? 1 : 0);
    }

    // ---- attention: squish = tanh(h2 @ W_att + b_att)   (into xp buffer) ----
    float* squish = xp;
    launch_gemm(hseq2, HID, watt, HID, b_att, nullptr, 0, squish, nullptr, HID,
                TT * BB, HID, HID, HID, 1, 4, stream);
    hipLaunchKernelGGL(att_score, dim3(TT * BB / 8), dim3(256), 0, stream,
                       squish, v_att, scores);
    hipLaunchKernelGGL(att_pool, dim3(BB), dim3(128), 0, stream,
                       scores, hseq2, pooled);

    // ---- head: hdn = pooled @ W1^T + b1 ;  out = hdn @ W2^T + b2 ----
    launch_gemm(pooled, HID, w1b, HID, b1, nullptr, 0, nullptr, hdn, 64,
                BB, 64, HID, 64, 0, 4, stream);
    launch_gemm(hdn, 64, w2p, 64, b2p, nullptr, 0, (float*)d_out, nullptr, NCLS,
                BB, 16, 64, NCLS, 0, 1, stream);
}
